// MoE_43739946942728
// MI455X (gfx1250) — matrix operand order repaired and checked by reading, not compiled, not run on hardware
//
#include <hip/hip_runtime.h>
#include <hip/hip_bf16.h>
#include <math.h>

// ---------------- problem constants ----------------
constexpr int Bc   = 4;
constexpr int Sc   = 2048;
constexpr int Dc   = 1024;           // model dim (K of GEMM1, N of GEMM2)
constexpr int FFc  = 4096;           // ffn dim   (N of GEMM1, K of GEMM2)
constexpr int Ec   = 8;              // experts
constexpr int STOK = Bc * Sc;        // 8192 tokens
constexpr int CAP  = 1024;           // ceil(8192*1.0/8)

constexpr int OUT_LAUX = STOK * Dc;  // flat index of l_aux in d_out
constexpr int OUT_TOTAL = OUT_LAUX + 1 + Ec;

// ---------------- workspace layout (bytes) ----------------
constexpr size_t XD_OFF    = 0;                                   // bf16 [8192,1024]
constexpr size_t H_OFF     = XD_OFF + (size_t)STOK * Dc * 2;      // bf16 [8192,4096]
constexpr size_t EIDX_OFF  = H_OFF + (size_t)STOK * FFc * 2;      // int  [8192]
constexpr size_t GATE_OFF  = EIDX_OFF + STOK * 4;                 // f32  [8192]
constexpr size_t NSEL_OFF  = GATE_OFF + STOK * 4;                 // f32  [8192]
constexpr size_t SLOT_OFF  = NSEL_OFF + STOK * 4;                 // int  [8192] (E*CAP)
constexpr size_t CTOK_OFF  = SLOT_OFF + STOK * 4;                 // int  [8,8192]
constexpr size_t MESUM_OFF = CTOK_OFF + (size_t)Ec * STOK * 4;    // f32  [8]
constexpr size_t CNT_OFF   = MESUM_OFF + 32;                      // int  [8]

// ---------------- WMMA types ----------------
typedef __attribute__((ext_vector_type(16))) __bf16 v16bf;
typedef __attribute__((ext_vector_type(8)))  float  v8f;

// A-fragment: lane row = lane&15; K chunks at (lane>>4)*8 and +16 (ISA 7.12.2)
__device__ inline v16bf load_fragA(const __bf16* rowbase, int kbase) {
  union { v16bf v; uint4 q[2]; } u;
  u.q[0] = *(const uint4*)(rowbase + kbase);
  u.q[1] = *(const uint4*)(rowbase + kbase + 16);
  return u.v;
}
// B-fragment: lane col = lane&15; K chunks at (lane>>4)*8 and +16
__device__ inline v16bf load_fragB(const __bf16* colbase, int lanehi) {
  union { v16bf v; uint4 q[2]; } u;
  const __bf16* p = colbase + lanehi * 8;
  u.q[0] = *(const uint4*)(p);
  u.q[1] = *(const uint4*)(p + 16);
  return u.v;
}

// pack two f32 into one dword of two bf16 (low = first); pinned to the
// dual-source v_cvt_pk_bf16_f32 so the compiler cannot re-pair operands.
__device__ inline unsigned pack_bf16(float a, float b) {
  unsigned r;
  asm("v_cvt_pk_bf16_f32 %0, %1, %2" : "=v"(r) : "v"(a), "v"(b));
  return r;
}
// plain conversion-based pack for non-critical kernels
__device__ inline unsigned pack_bf16_c(float a, float b) {
  union { __bf16 h[2]; unsigned u; } p;
  p.h[0] = (__bf16)a; p.h[1] = (__bf16)b;
  return p.u;
}

// 32B async copy global -> LDS (two b128 ops; INST_OFFSET applies to both
// the LDS and global address per ISA pseudocode). Tracked by ASYNCcnt.
__device__ inline void async_copy32(unsigned lds_addr, const void* gptr) {
  unsigned long long ga = (unsigned long long)(uintptr_t)gptr;
  asm volatile("global_load_async_to_lds_b128 %0, %1, off\n\t"
               "global_load_async_to_lds_b128 %0, %1, off offset:16"
               :: "v"(lds_addr), "v"(ga) : "memory");
}
__device__ inline void wait_async0() {
  asm volatile("s_wait_asynccnt 0x0" ::: "memory");
}

// ---------------- kernel 0: zero out + tiny accumulators ----------------
__global__ __launch_bounds__(256)
void moe_zero_kernel(float* out, float* me_sum, int* cnt) {
  size_t gid = (size_t)blockIdx.x * 256 + threadIdx.x;
  for (size_t i = gid; i < (size_t)OUT_TOTAL; i += 256ull * 8192ull) out[i] = 0.0f;
  if (gid < Ec) { me_sum[gid] = 0.0f; cnt[gid] = 0; }
}

// ---------------- kernel 1: gating (one wave32 per token) ----------------
__global__ __launch_bounds__(256)
void moe_gate_kernel(const float* __restrict__ x, const float* __restrict__ wg,
                     const float* __restrict__ rts, int* __restrict__ eidx,
                     float* __restrict__ gval, float* __restrict__ nsel,
                     float* me_sum, int* cnt) {
  __shared__ float wgs[Ec * Dc];               // transposed [e][d] -> conflict-free
  const int tid = threadIdx.x;
  for (int i = tid; i < Ec * Dc; i += 256) {
    int d = i >> 3, e = i & 7;
    wgs[e * Dc + d] = wg[i];
  }
  __syncthreads();
  const int wid = tid >> 5, lane = tid & 31;
  const int t = blockIdx.x * 8 + wid;
  float acc[Ec];
#pragma unroll
  for (int e = 0; e < Ec; ++e) acc[e] = 0.0f;
  const float* xr = x + (size_t)t * Dc;
  for (int d = lane; d < Dc; d += 32) {
    float xv = xr[d];
#pragma unroll
    for (int e = 0; e < Ec; ++e) acc[e] += xv * wgs[e * Dc + d];
  }
#pragma unroll
  for (int e = 0; e < Ec; ++e)
#pragma unroll
    for (int off = 16; off >= 1; off >>= 1)
      acc[e] += __shfl_xor(acc[e], off, 32);
  if (lane == 0) {
    float m = acc[0]; int idx = 0;
#pragma unroll
    for (int e = 1; e < Ec; ++e) if (acc[e] > m) { m = acc[e]; idx = e; }
    float g[Ec], ssum = 0.0f;
#pragma unroll
    for (int e = 0; e < Ec; ++e) { g[e] = expf(acc[e] - m); ssum += g[e]; }
    float inv = 1.0f / ssum;
#pragma unroll
    for (int e = 0; e < Ec; ++e) { g[e] *= inv; atomicAdd(&me_sum[e], g[e]); }
    atomicAdd(&cnt[idx], 1);
    eidx[t] = idx;
    gval[t] = g[idx];
    nsel[t] = rts[(size_t)t * Ec + idx];
  }
}

// ---------------- kernel 2: l_aux + exp_counts ----------------
__global__ void moe_finalize_kernel(const float* me_sum, const int* cnt, float* out) {
  const int tid = threadIdx.x;
  float p = 0.0f;
  if (tid < Ec) p = (me_sum[tid] / (float)STOK) * ((float)cnt[tid] / (float)STOK);
#pragma unroll
  for (int off = 16; off >= 1; off >>= 1) p += __shfl_xor(p, off, 32);
  if (tid == 0) out[OUT_LAUX] = p * (float)Ec;
  if (tid < Ec) ((int*)out)[OUT_LAUX + 1 + tid] = cnt[tid];
}

// ---------------- kernel 3: per-expert RTS selection + slot assignment ----------
__global__ __launch_bounds__(1024)
void moe_rts_kernel(const int* __restrict__ eidx, const float* __restrict__ nsel,
                    const int* __restrict__ cnt, int* __restrict__ ctok_all,
                    int* __restrict__ slot_token) {
  const int e = blockIdx.x, tid = threadIdx.x;
  __shared__ float nsh[STOK];      // 32 KB
  __shared__ int   scan[1024];
  __shared__ int   base_sh;
  slot_token[e * CAP + tid] = -1;  // blockDim == CAP
  if (tid == 0) base_sh = 0;
  __syncthreads();
  int* ctok = ctok_all + (size_t)e * STOK;
  // pass 1: stable compaction of this expert's tokens (token order)
  for (int ch = 0; ch < STOK / 1024; ++ch) {
    const int t = ch * 1024 + tid;
    const int pred = (eidx[t] == e) ? 1 : 0;
    scan[tid] = pred; __syncthreads();
    for (int off = 1; off < 1024; off <<= 1) {
      int v = (tid >= off) ? scan[tid - off] : 0;
      __syncthreads();
      scan[tid] += v; __syncthreads();
    }
    const int pos = base_sh + scan[tid] - pred;
    if (pred) { ctok[pos] = t; nsh[pos] = nsel[t]; }
    __syncthreads();
    if (tid == 0) base_sh += scan[1023];
    __syncthreads();
  }
  const int count = cnt[e];
  float thr = -1.0f;
  if (count > CAP) {
    for (int i = tid; i < STOK; i += 1024) if (i >= count) nsh[i] = -1.0f;
    __syncthreads();
    // bitonic sort, descending
    for (int k = 2; k <= STOK; k <<= 1)
      for (int j = k >> 1; j > 0; j >>= 1) {
        for (int i = tid; i < STOK; i += 1024) {
          const int ixj = i ^ j;
          if (ixj > i) {
            const bool desc = ((i & k) == 0);
            float a = nsh[i], b = nsh[ixj];
            if (desc ? (a < b) : (a > b)) { nsh[i] = b; nsh[ixj] = a; }
          }
        }
        __syncthreads();
      }
    thr = nsh[CAP - 1];   // C-th largest noise among assigned tokens
    __syncthreads();
  }
  // pass 2: stable compaction of kept tokens -> capacity slot index
  if (tid == 0) base_sh = 0;
  __syncthreads();
  const int chunks = (count + 1023) >> 10;
  for (int ch = 0; ch < chunks; ++ch) {
    const int i = ch * 1024 + tid;
    int pred = 0, tok = -1;
    if (i < count) {
      tok = ctok[i];
      pred = (nsel[tok] >= thr) ? 1 : 0;
    }
    scan[tid] = pred; __syncthreads();
    for (int off = 1; off < 1024; off <<= 1) {
      int v = (tid >= off) ? scan[tid - off] : 0;
      __syncthreads();
      scan[tid] += v; __syncthreads();
    }
    const int loc = base_sh + scan[tid] - pred;
    if (pred && loc < CAP) slot_token[e * CAP + loc] = tok;
    __syncthreads();
    if (tid == 0) base_sh += scan[1023];
    __syncthreads();
  }
}

// ---------------- kernel 4: dispatch-gather to bf16 [E*C, D] ----------------
__global__ __launch_bounds__(256)
void moe_dispatch_kernel(const float* __restrict__ x, const int* __restrict__ slot_token,
                         __bf16* __restrict__ Xd) {
  const int slot = blockIdx.x, tid = threadIdx.x;
  const int tok = slot_token[slot];
  unsigned* dst = (unsigned*)(Xd + (size_t)slot * Dc) + tid * 2;
  if (tok >= 0) {
    const float4 v = *(const float4*)(x + (size_t)tok * Dc + tid * 4);
    dst[0] = pack_bf16_c(v.x, v.y);
    dst[1] = pack_bf16_c(v.z, v.w);
  } else {
    dst[0] = 0u;
    dst[1] = 0u;
  }
}

// ---------------- fused WMMA GEMM (both FFN layers) -------------------------
// GELU=true : Hout[m, NB] = gelu(A@Bw + bias)        (bf16 out)
// GELU=false: out[token]  = gval * (A@Bw + bias)     (gated scatter via slot_token)
// A: [8192, KDIM] bf16 row-major; Bw: per-expert [KDIM, NB] fp32 row-major.
// Double-buffered LDS; A tile staged with GLOBAL_LOAD_ASYNC_TO_LDS_B128
// (ASYNCcnt); B tile register-prefetched as (k, k+1) row pairs, packed with
// pinned v_cvt_pk_bf16_f32 and stored as dwords (ds_store_b32/2addr).
template <int KDIM, int NB, bool GELU>
__global__ __launch_bounds__(256)
void moe_gemm_kernel(const __bf16* __restrict__ A, const float* __restrict__ Bw,
                     const float* __restrict__ bias, __bf16* __restrict__ Hout,
                     const int* __restrict__ slot_token, const float* __restrict__ gval,
                     float* __restrict__ out) {
  __shared__ __bf16 As[2][128 * 32];   // [row][k]
  __shared__ __bf16 Bs[2][128 * 32];   // [n][k] (transposed while staging)
  const int gm = blockIdx.x * 128;
  const int n0 = blockIdx.y * 128;
  const int e  = gm >> 10;                      // 1024 rows per expert
  const float* Be = Bw + (size_t)e * KDIM * NB;
  const int tid = threadIdx.x;
  const int wid = tid >> 5, lane = tid & 31;
  const int wm = wid >> 1, wn = wid & 1;
  const int arow = tid >> 1, ahalf = tid & 1;   // A staging: 32B per thread
  const int bkp = (tid & 15) * 2;               // B staging: k pair (even k)
  const int bng = (tid >> 4) * 8;               //            8 consecutive n
  const int kbase = (lane >> 4) * 8, lanehi = lane >> 4, lrow = lane & 15;

  const __bf16* aSrc = A + (size_t)(gm + arow) * KDIM + ahalf * 16;
  const float*  bSrc = Be + (size_t)bkp * NB + n0 + bng;
  const int bdi = bng * 16 + (bkp >> 1);        // dword index into Bs buffer
  const unsigned ldsA[2] = {
    (unsigned)(uintptr_t)&As[0][arow * 32 + ahalf * 16],
    (unsigned)(uintptr_t)&As[1][arow * 32 + ahalf * 16]
  };

  // ---- prologue: stage kk = 0 into buffer 0 ----
  {
    async_copy32(ldsA[0], aSrc);
    float4 r0a = *(const float4*)(bSrc);
    float4 r0b = *(const float4*)(bSrc + 4);
    float4 r1a = *(const float4*)(bSrc + NB);
    float4 r1b = *(const float4*)(bSrc + NB + 4);
    unsigned* bd = (unsigned*)&Bs[0][0];
    bd[bdi + 0 * 16] = pack_bf16(r0a.x, r1a.x);
    bd[bdi + 1 * 16] = pack_bf16(r0a.y, r1a.y);
    bd[bdi + 2 * 16] = pack_bf16(r0a.z, r1a.z);
    bd[bdi + 3 * 16] = pack_bf16(r0a.w, r1a.w);
    bd[bdi + 4 * 16] = pack_bf16(r0b.x, r1b.x);
    bd[bdi + 5 * 16] = pack_bf16(r0b.y, r1b.y);
    bd[bdi + 6 * 16] = pack_bf16(r0b.z, r1b.z);
    bd[bdi + 7 * 16] = pack_bf16(r0b.w, r1b.w);
    wait_async0();
    __syncthreads();
  }

  v8f acc[2][4] = {};
  float4 r0a, r0b, r1a, r1b;
  int cur = 0;
  for (int kk = 0; kk < KDIM; kk += 32) {
    const bool pf = (kk + 32 < KDIM);
    if (pf) {
      // async-stage next A tile straight into the other LDS buffer
      async_copy32(ldsA[cur ^ 1], aSrc + kk + 32);
      // register-prefetch next B tile (rows bkp, bkp+1)
      const float* bs = bSrc + (size_t)(kk + 32) * NB;
      r0a = *(const float4*)(bs);
      r0b = *(const float4*)(bs + 4);
      r1a = *(const float4*)(bs + NB);
      r1b = *(const float4*)(bs + NB + 4);
    }
    // compute current buffer
    v16bf af[2], bf[4];
#pragma unroll
    for (int mi = 0; mi < 2; ++mi)
      af[mi] = load_fragA(&As[cur][(wm * 32 + mi * 16 + lrow) * 32], kbase);
#pragma unroll
    for (int ni = 0; ni < 4; ++ni)
      bf[ni] = load_fragB(&Bs[cur][(wn * 64 + ni * 16 + lrow) * 32], lanehi);
#pragma unroll
    for (int mi = 0; mi < 2; ++mi)
#pragma unroll
      for (int ni = 0; ni < 4; ++ni)
        acc[mi][ni] = __builtin_amdgcn_wmma_f32_16x16x32_bf16(
            false, af[mi], false, bf[ni], (short)0, acc[mi][ni], false, false);
    if (pf) {
      // pack + store prefetched B into the other buffer
      unsigned* bd = (unsigned*)&Bs[cur ^ 1][0];
      bd[bdi + 0 * 16] = pack_bf16(r0a.x, r1a.x);
      bd[bdi + 1 * 16] = pack_bf16(r0a.y, r1a.y);
      bd[bdi + 2 * 16] = pack_bf16(r0a.z, r1a.z);
      bd[bdi + 3 * 16] = pack_bf16(r0a.w, r1a.w);
      bd[bdi + 4 * 16] = pack_bf16(r0b.x, r1b.x);
      bd[bdi + 5 * 16] = pack_bf16(r0b.y, r1b.y);
      bd[bdi + 6 * 16] = pack_bf16(r0b.z, r1b.z);
      bd[bdi + 7 * 16] = pack_bf16(r0b.w, r1b.w);
      wait_async0();
    }
    __syncthreads();
    cur ^= 1;
  }

  // ---- epilogue ----
  const int lhi = (lane >> 4) * 8;
  if (GELU) {
    const float* be = bias + (size_t)e * NB;
#pragma unroll
    for (int mi = 0; mi < 2; ++mi)
#pragma unroll
      for (int ni = 0; ni < 4; ++ni) {
        const int c = wn * 64 + ni * 16 + lrow;
        const float b = be[n0 + c];
#pragma unroll
        for (int j = 0; j < 8; ++j) {
          const int r = wm * 32 + mi * 16 + lhi + j;
          float v = acc[mi][ni][j] + b;
          v = 0.5f * v * (1.0f + erff(v * 0.70710678118654752f));
          Hout[(size_t)(gm + r) * NB + n0 + c] = (__bf16)v;
        }
      }
  } else {
    const float* be = bias + (size_t)e * NB;
#pragma unroll
    for (int mi = 0; mi < 2; ++mi)
#pragma unroll
      for (int j = 0; j < 8; ++j) {
        const int r = wm * 32 + mi * 16 + lhi + j;
        const int tok = slot_token[gm + r];
        if (tok < 0) continue;
        const float g = gval[tok];
        float* orow = out + (size_t)tok * NB;
#pragma unroll
        for (int ni = 0; ni < 4; ++ni) {
          const int c = wn * 64 + ni * 16 + lrow;
          orow[n0 + c] = g * (acc[mi][ni][j] + be[n0 + c]);
        }
      }
  }
}

// ---------------- launch ----------------
extern "C" void kernel_launch(void* const* d_in, const int* in_sizes, int n_in,
                              void* d_out, int out_size, void* d_ws, size_t ws_size,
                              hipStream_t stream) {
  const float* x   = (const float*)d_in[0];
  const float* wg  = (const float*)d_in[1];
  const float* w1  = (const float*)d_in[2];
  const float* b1  = (const float*)d_in[3];
  const float* w2  = (const float*)d_in[4];
  const float* b2  = (const float*)d_in[5];
  const float* rts = (const float*)d_in[6];
  float* out = (float*)d_out;
  char* ws = (char*)d_ws;

  __bf16* Xd      = (__bf16*)(ws + XD_OFF);
  __bf16* H       = (__bf16*)(ws + H_OFF);
  int*    eidx    = (int*)   (ws + EIDX_OFF);
  float*  gval    = (float*) (ws + GATE_OFF);
  float*  nsel    = (float*) (ws + NSEL_OFF);
  int*    slot    = (int*)   (ws + SLOT_OFF);
  int*    ctok    = (int*)   (ws + CTOK_OFF);
  float*  me_sum  = (float*) (ws + MESUM_OFF);
  int*    cnt     = (int*)   (ws + CNT_OFF);

  moe_zero_kernel<<<8192, 256, 0, stream>>>(out, me_sum, cnt);
  moe_gate_kernel<<<STOK / 8, 256, 0, stream>>>(x, wg, rts, eidx, gval, nsel, me_sum, cnt);
  moe_finalize_kernel<<<1, 32, 0, stream>>>(me_sum, cnt, out);
  moe_rts_kernel<<<Ec, 1024, 0, stream>>>(eidx, nsel, cnt, ctok, slot);
  moe_dispatch_kernel<<<STOK, 256, 0, stream>>>(x, slot, Xd);
  // GEMM1: [8192,1024] @ [1024,4096] -> GELU -> H (bf16)
  moe_gemm_kernel<Dc, FFc, true><<<dim3(STOK / 128, FFc / 128), 256, 0, stream>>>(
      Xd, w1, b1, H, nullptr, nullptr, nullptr);
  // GEMM2: [8192,4096] @ [4096,1024] + b2, gated scatter to out
  moe_gemm_kernel<FFc, Dc, false><<<dim3(STOK / 128, Dc / 128), 256, 0, stream>>>(
      H, w2, b2, nullptr, slot, gval, out);
}
